// LSTMOld_40853728919900
// MI455X (gfx1250) — matrix operand order repaired and checked by reading, not compiled, not run on hardware
//
#include <hip/hip_runtime.h>
#include <math.h>

// ---------------- problem constants ----------------
#define TT 512
#define BB 32
#define II 1024
#define HH 1024
#define G4 4096   // 4*H

typedef __attribute__((ext_vector_type(8)))  __bf16 v8bf;
typedef __attribute__((ext_vector_type(16))) __bf16 v16bf;
typedef __attribute__((ext_vector_type(8)))  float  v8f;

// ---------------- fp32 -> bf16 (RNE) ----------------
__device__ __forceinline__ unsigned short f32_to_bf16(float f) {
    unsigned int u = __float_as_uint(f);
    return (unsigned short)((u + 0x7FFFu + ((u >> 16) & 1u)) >> 16);
}

__global__ void k_cvt_bf16(const float* __restrict__ src,
                           unsigned short* __restrict__ dst, int n) {
    int i = blockIdx.x * blockDim.x + threadIdx.x;
    if (i < n) dst[i] = f32_to_bf16(src[i]);
}

__global__ void k_init_bar(unsigned* bar) { bar[0] = 0u; bar[1] = 0u; }

// ---------------- WMMA fragment loaders ----------------
// A operand: 16x32 bf16 tile from a row-major [M x K] matrix (global or LDS;
// address space inferred after inlining).  ISA 7.12.2: lanes 0-15 hold rows
// 0-15 with K = {0..7,16..23}; lanes 16-31 hold K = {8..15,24..31}.
__device__ __forceinline__ v16bf load_frag_a(const __bf16* m,
                                             int row0, int ld, int kb) {
    int lane = threadIdx.x & 31;
    int r  = lane & 15;
    int kh = lane >> 4;                       // 0 or 1
    const __bf16* p = m + (size_t)(row0 + r) * ld + kb + kh * 8;
    v8bf lo = *reinterpret_cast<const v8bf*>(p);        // K = kh*8 + 0..7
    v8bf hi = *reinterpret_cast<const v8bf*>(p + 16);   // K = 16 + kh*8 + 0..7
    return __builtin_shufflevector(lo, hi, 0,1,2,3,4,5,6,7,8,9,10,11,12,13,14,15);
}

// B operand: 32x16 bf16 tile (K x N) sourced from a row-major [N x K] matrix
// (gates = act @ W^T).  Lane n holds column n; lanes 0-15 carry K={0..7,16..23},
// lanes 16-31 carry K={8..15,24..31}.
__device__ __forceinline__ v16bf load_frag_b(const __bf16* m,
                                             int col0, int ld, int kb) {
    int lane = threadIdx.x & 31;
    int n  = lane & 15;
    int kh = lane >> 4;
    const __bf16* p = m + (size_t)(col0 + n) * ld + kb + kh * 8;
    v8bf lo = *reinterpret_cast<const v8bf*>(p);
    v8bf hi = *reinterpret_cast<const v8bf*>(p + 16);
    return __builtin_shufflevector(lo, hi, 0,1,2,3,4,5,6,7,8,9,10,11,12,13,14,15);
}

#define WMMA_BF16(A, B, C) \
    __builtin_amdgcn_wmma_f32_16x16x32_bf16(false, (A), false, (B), (short)0, (C), false, false)

// ---------------- Phase 1: xg = x @ w_ih^T + (b_ih + b_hh) ----------------
// LDS-tiled GEMM: block tile 128(M) x 128(N), K-step 32.  8 waves in a 4x2
// grid; each wave computes 32x64 (2x4 WMMA subtiles).  Operand traffic from
// L2 drops 4x vs direct-from-global (x read 32x, w_ih read 128x).
__global__ void __launch_bounds__(256)
k_gemm_xg(const unsigned short* __restrict__ xbf,
          const unsigned short* __restrict__ wbf,
          const float* __restrict__ b_ih, const float* __restrict__ b_hh,
          float* __restrict__ xg) {
    __shared__ unsigned short lds_a_raw[128 * 32];   // 8 KB
    __shared__ unsigned short lds_b_raw[128 * 32];   // 8 KB
    __bf16* lds_a = reinterpret_cast<__bf16*>(lds_a_raw);
    __bf16* lds_b = reinterpret_cast<__bf16*>(lds_b_raw);

    const int tid = threadIdx.x;
    const int MB  = (TT * BB) / 128;          // 128 M-blocks
    const int mb  = blockIdx.x % MB;
    const int nb  = blockIdx.x / MB;          // 0..31
    const int w   = tid >> 5;
    const int wr  = w & 3;                    // wave M row (0..3)
    const int wc  = w >> 2;                   // wave N col (0..1)
    const int mbase = wr * 32;
    const int nbase = wc * 64;

    const __bf16* A = reinterpret_cast<const __bf16*>(xbf) + (size_t)mb * 128 * II;
    const __bf16* W = reinterpret_cast<const __bf16*>(wbf) + (size_t)nb * 128 * II;

    // copy indices: each thread moves 32B (16 bf16) of A and of B per K-step
    const int crow = tid >> 1;                // 0..127
    const int cseg = (tid & 1) * 16;          // halves offset (0 or 16)

    v8f acc[2][4];
    for (int mt = 0; mt < 2; ++mt)
        for (int nt = 0; nt < 4; ++nt) acc[mt][nt] = (v8f){};

    for (int kb = 0; kb < II; kb += 32) {
        // ---- global -> LDS (A: 8KB, B: 8KB) ----
        {
            const uint4* sa = reinterpret_cast<const uint4*>(A + (size_t)crow * II + kb + cseg);
            uint4* da = reinterpret_cast<uint4*>(lds_a + crow * 32 + cseg);
            da[0] = sa[0]; da[1] = sa[1];
            const uint4* sb = reinterpret_cast<const uint4*>(W + (size_t)crow * II + kb + cseg);
            uint4* db = reinterpret_cast<uint4*>(lds_b + crow * 32 + cseg);
            db[0] = sb[0]; db[1] = sb[1];
        }
        __syncthreads();

        v16bf a0 = load_frag_a(lds_a, mbase,      32, 0);
        v16bf a1 = load_frag_a(lds_a, mbase + 16, 32, 0);
        v16bf b0 = load_frag_b(lds_b, nbase,      32, 0);
        v16bf b1 = load_frag_b(lds_b, nbase + 16, 32, 0);
        v16bf b2 = load_frag_b(lds_b, nbase + 32, 32, 0);
        v16bf b3 = load_frag_b(lds_b, nbase + 48, 32, 0);

        acc[0][0] = WMMA_BF16(a0, b0, acc[0][0]);
        acc[0][1] = WMMA_BF16(a0, b1, acc[0][1]);
        acc[0][2] = WMMA_BF16(a0, b2, acc[0][2]);
        acc[0][3] = WMMA_BF16(a0, b3, acc[0][3]);
        acc[1][0] = WMMA_BF16(a1, b0, acc[1][0]);
        acc[1][1] = WMMA_BF16(a1, b1, acc[1][1]);
        acc[1][2] = WMMA_BF16(a1, b2, acc[1][2]);
        acc[1][3] = WMMA_BF16(a1, b3, acc[1][3]);
        __syncthreads();
    }

    const int lane = tid & 31;
    const int n    = lane & 15;
    const int mofs = (lane >> 4) * 8;         // C/D layout: VGPR v -> M = v (+8)
    const size_t row0 = (size_t)mb * 128 + mbase;
    const int    col0 = nb * 128 + nbase;
    for (int nt = 0; nt < 4; ++nt) {
        const int c = col0 + nt * 16 + n;
        const float bias = b_ih[c] + b_hh[c];
        for (int mt = 0; mt < 2; ++mt) {
            for (int v = 0; v < 8; ++v) {
                int m = mt * 16 + v + mofs;
                xg[(row0 + m) * G4 + c] = acc[mt][nt][v] + bias;
            }
        }
    }
}

// ---------------- Phase 2: persistent recurrent scan ----------------
// 32 blocks x 128 threads.  Block bk owns hidden columns [bk*32, bk*32+32).
// Wave g (0..3) computes gate g's [32 batch x 32 col] tile via WMMA, with h
// staged once per step into LDS (shared by all 4 waves).  c lives in LDS for
// the whole kernel; bf16 h ping-pongs through global ws.
__global__ void __launch_bounds__(128)
k_lstm_scan(const float* __restrict__ xg,
            const unsigned short* __restrict__ whh,
            unsigned short* __restrict__ hbuf,          // [2][B*H] bf16
            const float* __restrict__ c0,
            float* __restrict__ out,
            unsigned* __restrict__ bar) {
    __shared__ unsigned short h_raw[BB * HH];  // 64 KB staged h (of 320 KB WGP LDS)
    __shared__ float gate_res[4 * 32 * 32];    // 16 KB
    __shared__ float c_sl[32 * 32];            //  4 KB
    __bf16* h_lds = reinterpret_cast<__bf16*>(h_raw);

    const int bk   = blockIdx.x;
    const int tid  = threadIdx.x;
    const int wave = tid >> 5;                 // = gate index
    const int col0 = bk * 32;

    for (int e = tid; e < 1024; e += 128) {
        int b = e >> 5, j = e & 31;
        c_sl[e] = c0[b * HH + col0 + j];
    }

    const __bf16* W = reinterpret_cast<const __bf16*>(whh) +
                      (size_t)(wave * HH + col0) * HH;
    unsigned* cnt = bar;
    unsigned* gen = bar + 1;

    for (int t = 0; t < TT; ++t) {
        // ---- stage h(t) into LDS: 64KB, 128 threads x 512B ----
        {
            const uint4* src = reinterpret_cast<const uint4*>(
                hbuf + (size_t)(t & 1) * (BB * HH));
            uint4* dst = reinterpret_cast<uint4*>(h_raw);
            for (int i = tid; i < (BB * HH) / 8; i += 128) dst[i] = src[i];
        }

        // warm xg lines for the combine phase while the GEMM runs
        for (int e = tid; e < 1024; e += 128) {
            int b = e >> 5, j = e & 31;
            __builtin_prefetch(&xg[((size_t)t * BB + b) * G4 + col0 + j], 0, 1);
        }
        __syncthreads();

        v8f acc00 = {}, acc01 = {}, acc10 = {}, acc11 = {};
        for (int kb = 0; kb < HH; kb += 32) {
            v16bf a0 = load_frag_a(h_lds, 0,  HH, kb);
            v16bf a1 = load_frag_a(h_lds, 16, HH, kb);
            v16bf b0 = load_frag_b(W,  0,  HH, kb);
            v16bf b1 = load_frag_b(W,  16, HH, kb);
            acc00 = WMMA_BF16(a0, b0, acc00);
            acc01 = WMMA_BF16(a0, b1, acc01);
            acc10 = WMMA_BF16(a1, b0, acc10);
            acc11 = WMMA_BF16(a1, b1, acc11);
        }
        {
            int lane = tid & 31;
            int n    = lane & 15;
            int mofs = (lane >> 4) * 8;
            float* gr = &gate_res[wave * 1024];
            for (int v = 0; v < 8; ++v) {
                int m = v + mofs;
                gr[(m)      * 32 + n]      = acc00[v];
                gr[(m)      * 32 + 16 + n] = acc01[v];
                gr[(16 + m) * 32 + n]      = acc10[v];
                gr[(16 + m) * 32 + 16 + n] = acc11[v];
            }
        }
        __syncthreads();

        unsigned short* Hd = hbuf + (size_t)((t + 1) & 1) * (BB * HH);
        for (int e = tid; e < 1024; e += 128) {
            int b = e >> 5, j = e & 31;
            size_t xb = ((size_t)t * BB + b) * (size_t)G4 + col0 + j;
            float gi = xg[xb + 0 * HH] + gate_res[0 * 1024 + e];
            float gf = xg[xb + 1 * HH] + gate_res[1 * 1024 + e];
            float gg = xg[xb + 2 * HH] + gate_res[2 * 1024 + e];
            float go = xg[xb + 3 * HH] + gate_res[3 * 1024 + e];
            gi = 1.0f / (1.0f + expf(-gi));
            gf = 1.0f / (1.0f + expf(-gf));
            go = 1.0f / (1.0f + expf(-go));
            gg = tanhf(gg);
            float c = gf * c_sl[e] + gi * gg;
            float h = go * tanhf(c);
            c_sl[e] = c;
            out[(size_t)t * (BB * HH) + b * HH + col0 + j] = h;
            Hd[b * HH + col0 + j] = f32_to_bf16(h);
            if (t == TT - 1) {
                out[(size_t)TT * BB * HH + b * HH + col0 + j]           = h;  // hT
                out[(size_t)TT * BB * HH + BB * HH + b * HH + col0 + j] = c;  // cT
            }
        }

        // ---- grid-wide barrier (monotonic generation, self-resetting) ----
        __threadfence();
        __syncthreads();
        if (tid == 0) {
            unsigned g0 = __hip_atomic_load(gen, __ATOMIC_RELAXED,
                                            __HIP_MEMORY_SCOPE_AGENT);
            unsigned prev = __hip_atomic_fetch_add(cnt, 1u, __ATOMIC_ACQ_REL,
                                                   __HIP_MEMORY_SCOPE_AGENT);
            if (prev == gridDim.x - 1) {
                __hip_atomic_store(cnt, 0u, __ATOMIC_RELAXED,
                                   __HIP_MEMORY_SCOPE_AGENT);
                __hip_atomic_fetch_add(gen, 1u, __ATOMIC_ACQ_REL,
                                       __HIP_MEMORY_SCOPE_AGENT);
            } else {
                while (__hip_atomic_load(gen, __ATOMIC_ACQUIRE,
                                         __HIP_MEMORY_SCOPE_AGENT) == g0) {
                    __builtin_amdgcn_s_sleep(1);
                }
            }
        }
        __syncthreads();
    }
}

// ---------------- workspace layout (bytes) ----------------
static const size_t OFF_XG  = 0;                                   // fp32 [T*B,4H]
static const size_t SZ_XG   = (size_t)TT * BB * G4 * 4;            // 256 MiB
static const size_t OFF_XBF = OFF_XG + SZ_XG;                      // bf16 x
static const size_t SZ_XBF  = (size_t)TT * BB * II * 2;
static const size_t OFF_WIH = OFF_XBF + SZ_XBF;
static const size_t SZ_WIH  = (size_t)G4 * II * 2;
static const size_t OFF_WHH = OFF_WIH + SZ_WIH;
static const size_t SZ_WHH  = (size_t)G4 * HH * 2;
static const size_t OFF_HB  = OFF_WHH + SZ_WHH;                    // bf16 h x2
static const size_t SZ_HB   = (size_t)2 * BB * HH * 2;
static const size_t OFF_BAR = OFF_HB + SZ_HB;

extern "C" void kernel_launch(void* const* d_in, const int* in_sizes, int n_in,
                              void* d_out, int out_size, void* d_ws, size_t ws_size,
                              hipStream_t stream) {
    const float* x    = (const float*)d_in[0];
    const float* h0   = (const float*)d_in[1];
    const float* c0   = (const float*)d_in[2];
    const float* w_ih = (const float*)d_in[3];
    const float* w_hh = (const float*)d_in[4];
    const float* b_ih = (const float*)d_in[5];
    const float* b_hh = (const float*)d_in[6];
    float* out = (float*)d_out;

    char* ws = (char*)d_ws;
    float*          xg   = (float*)(ws + OFF_XG);
    unsigned short* xbf  = (unsigned short*)(ws + OFF_XBF);
    unsigned short* wihb = (unsigned short*)(ws + OFF_WIH);
    unsigned short* whhb = (unsigned short*)(ws + OFF_WHH);
    unsigned short* hbuf = (unsigned short*)(ws + OFF_HB);
    unsigned*       bar  = (unsigned*)(ws + OFF_BAR);

    // bf16 conversions (one pass; weights then live in the 192MB L2)
    int nx = TT * BB * II;
    k_cvt_bf16<<<(nx + 255) / 256, 256, 0, stream>>>(x, xbf, nx);
    int nw = G4 * II;
    k_cvt_bf16<<<(nw + 255) / 256, 256, 0, stream>>>(w_ih, wihb, nw);
    k_cvt_bf16<<<(nw + 255) / 256, 256, 0, stream>>>(w_hh, whhb, nw);
    int nh = BB * HH;
    k_cvt_bf16<<<(nh + 255) / 256, 256, 0, stream>>>(h0, hbuf, nh); // buffer 0
    k_init_bar<<<1, 1, 0, stream>>>(bar);

    // Phase 1: 128 x 32 = 4096 blocks of 128x128 output tile
    k_gemm_xg<<<4096, 256, 0, stream>>>(xbf, wihb, b_ih, b_hh, xg);

    // Phase 2: persistent scan, 32 blocks (one per 32-column slice of H)
    k_lstm_scan<<<32, 128, 0, stream>>>(xg, whhb, hbuf, c0, out, bar);
}
